// LTAE2d_34316788695975
// MI455X (gfx1250) — matrix operand order repaired and checked by reading, not compiled, not run on hardware
//
#include <hip/hip_runtime.h>

// ---------------- CDNA5 WMMA vector types ----------------
typedef __attribute__((ext_vector_type(16))) __bf16 v16bf;
typedef __attribute__((ext_vector_type(8)))  float  v8f;

// ---------------- problem constants ----------------
#define SZB    2
#define TLEN   24
#define LPIX   4096
#define CIN    128
#define DMODEL 256
#define NHEAD  16
#define DKQ    4
#define DHEAD  16
#define MLPOUT 128

// ---------------- workspace layout (bytes) ----------------
// B-fragment packing: frag = 32 lanes * 16 bf16 = 512 elems = 1 KB
#define WS_BIN_OFF   0u        // 64 frags (4 ktiles * 16 ntiles) * 1KB = 64 KB
#define WS_BK_OFF    65536u    // 32 frags (8 ktiles * 4  ntiles) * 1KB = 32 KB
#define WS_PROJT_OFF 98304u    // proj_w transposed f32: 256*128*4 = 128 KB
#define WS_PE_OFF    229376u   // PE table f32: 2*24*16*4 = 3 KB

// =====================================================================
// Prep kernel: pack weights into WMMA-B-fragment order (bf16), transpose
// proj_w, precompute the positional-encoding table.
//
// B-frag element map (32xK16 tile, wave32): lane L holds column N=(L&15);
// element i holds K = (L<16 ? i : 16+i).  Stored frag-major, lane-major so
// the GEMM loop loads 32 contiguous bytes per lane (2x global_load_b128).
// =====================================================================
__global__ void ltae_prep(const float* __restrict__ inconv_w,   // (256,128)
                          const float* __restrict__ fc1k_w,     // (64,256)
                          const float* __restrict__ proj_w,     // (128,256)
                          const int*   __restrict__ pos,        // (2,24)
                          __bf16* __restrict__ wsBin,
                          __bf16* __restrict__ wsBk,
                          float*  __restrict__ projT,
                          float*  __restrict__ pe)
{
    int g = blockIdx.x * blockDim.x + threadIdx.x;
    if (g < 32768) {                    // inconv_w fragments
        int fragIdx = g >> 9;           // /512
        int rem     = g & 511;
        int lane    = rem >> 4;
        int i       = rem & 15;
        int kt      = fragIdx >> 4;     // 0..3   (K base = 32*kt)
        int nt      = fragIdx & 15;     // 0..15  (N base = 16*nt)
        int K = kt * 32 + ((lane < 16) ? i : 16 + i);
        int N = nt * 16 + (lane & 15);
        wsBin[g] = (__bf16)inconv_w[N * CIN + K];
    } else if (g < 32768 + 16384) {     // fc1k_w fragments
        int q       = g - 32768;
        int fragIdx = q >> 9;
        int rem     = q & 511;
        int lane    = rem >> 4;
        int i       = rem & 15;
        int kt      = fragIdx >> 2;     // 0..7
        int nt      = fragIdx & 3;      // 0..3
        int K = kt * 32 + ((lane < 16) ? i : 16 + i);
        int N = nt * 16 + (lane & 15);
        wsBk[q] = (__bf16)fc1k_w[N * DMODEL + K];
    } else if (g < 32768 + 16384 + 32768) {   // proj_w transpose
        int q = g - 49152;
        int d = q >> 7;
        int m = q & 127;
        projT[q] = proj_w[m * DMODEL + d];
    } else if (g < 32768 + 16384 + 32768 + SZB * TLEN * DHEAD) {  // PE table
        int q  = g - 81920;
        int bt = q >> 4;
        int j  = q & 15;
        float p     = (float)pos[bt];
        float denom = powf(1000.0f, (float)(j >> 1) * (1.0f / 8.0f));
        float arg   = p / denom;
        pe[q] = ((j & 1) == 0) ? sinf(arg) : cosf(arg);
    }
}

// =====================================================================
// Fused LTAE kernel: one workgroup (128 threads = 4 wave32) per pixel.
// =====================================================================
__global__ __launch_bounds__(128) void ltae_fused(
    const float*         __restrict__ x,        // (48,4096,128)
    const unsigned char* __restrict__ pad_mask, // (2,24) bool
    const float*         __restrict__ ln_w,     // (128,)
    const float*         __restrict__ ln_b,     // (128,)
    const float*         __restrict__ inconv_b, // (256,)
    const float*         __restrict__ Qw,       // (16,4)
    const float*         __restrict__ fc1k_b,   // (64,)
    const float*         __restrict__ proj_b,   // (128,)
    const __bf16*        __restrict__ wsBin,
    const __bf16*        __restrict__ wsBk,
    const float*         __restrict__ projT,    // (256,128)
    const float*         __restrict__ pe,       // (2,24,16)
    float*               __restrict__ out)      // (2,4096,128)
{
    __shared__ float sX[TLEN * CIN];      // 12 KB: X tile, later K/attn/heads
    __shared__ float sE[TLEN * DMODEL];   // 24 KB: E = inconv(LN(x)) + PE (f32)
    __shared__ float sMu[TLEN];
    __shared__ float sRs[TLEN];

    float* sK     = sX;                   // 24*64  = 1536 floats
    float* sAttn  = sX + 1536;            // 24*16  = 384 floats
    float* sHeads = sX + 1920;            // 256 floats

    const int tid  = threadIdx.x;
    const int lane = tid & 31;            // wave32
    const int wv   = tid >> 5;            // 4 waves
    const int p    = blockIdx.x;          // pixel id
    const int b    = p >> 12;             // /4096
    const int l    = p & (LPIX - 1);

    // ---------- stage X (24x128) into LDS, coalesced float4 ----------
    for (int v = tid; v < TLEN * CIN / 4; v += 128) {
        int t  = v >> 5;                  // /32 float4 per row
        int c4 = v & 31;
        const float4* src = reinterpret_cast<const float4*>(
            x + ((size_t)(b * TLEN + t) * LPIX + l) * CIN);
        reinterpret_cast<float4*>(sX)[v] = src[c4];
    }
    __syncthreads();

    // ---------- LayerNorm over C (staggered to dodge bank conflicts) ----------
    if (tid < TLEN) {
        float s = 0.f, s2 = 0.f;
        for (int i = 0; i < CIN; ++i) {
            float vv = sX[tid * CIN + ((i + tid) & (CIN - 1))];
            s += vv; s2 += vv * vv;
        }
        float mu  = s * (1.0f / CIN);
        float var = s2 * (1.0f / CIN) - mu * mu;
        sMu[tid] = mu;
        sRs[tid] = rsqrtf(var + 1e-5f);
    }
    __syncthreads();
    for (int idx = tid; idx < TLEN * CIN; idx += 128) {
        int t = idx >> 7;
        int c = idx & (CIN - 1);
        sX[idx] = (sX[idx] - sMu[t]) * sRs[t] * ln_w[c] + ln_b[c];
    }
    __syncthreads();

    // ---------- GEMM1: E = Xln(24x128) @ Win^T(128x256), wave owns N-slice of 64 ----------
    v8f acc[2][4];
    {
        v8f z = {};
        #pragma unroll
        for (int mt = 0; mt < 2; ++mt)
            #pragma unroll
            for (int nt = 0; nt < 4; ++nt) acc[mt][nt] = z;
    }

    #pragma unroll
    for (int kt = 0; kt < 4; ++kt) {
        if (kt < 3)   // warm next K-slice of B frags (L2 resident) -> global_prefetch_b8
            __builtin_prefetch(wsBin + (size_t)((kt + 1) * 16 + wv * 4) * 512, 0, 0);

        // A fragments (16x32 bf16) for both M-tiles, built from LDS f32
        v16bf afr[2];
        #pragma unroll
        for (int mt = 0; mt < 2; ++mt) {
            int Mrow = (lane & 15) + 16 * mt;
            int hi   = (lane >= 16) ? 16 : 0;
            float f[16];
            if (Mrow < TLEN) {
                const float4* r0 = reinterpret_cast<const float4*>(&sX[Mrow * CIN + kt * 32 + hi]);
                const float4* r1 = reinterpret_cast<const float4*>(&sX[Mrow * CIN + kt * 32 + 8 + hi]);
                float4 u0 = r0[0], u1 = r0[1], u2 = r1[0], u3 = r1[1];
                f[0]=u0.x; f[1]=u0.y; f[2]=u0.z; f[3]=u0.w;
                f[4]=u1.x; f[5]=u1.y; f[6]=u1.z; f[7]=u1.w;
                f[8]=u2.x; f[9]=u2.y; f[10]=u2.z; f[11]=u2.w;
                f[12]=u3.x; f[13]=u3.y; f[14]=u3.z; f[15]=u3.w;
            } else {
                #pragma unroll
                for (int i = 0; i < 16; ++i) f[i] = 0.0f;
            }
            #pragma unroll
            for (int i = 0; i < 16; ++i) afr[mt][i] = (__bf16)f[i];
        }

        #pragma unroll
        for (int nt = 0; nt < 4; ++nt) {
            int fragIdx = kt * 16 + (wv * 4 + nt);
            v16bf bfr = *reinterpret_cast<const v16bf*>(wsBin + (size_t)fragIdx * 512 + lane * 16);
            acc[0][nt] = __builtin_amdgcn_wmma_f32_16x16x32_bf16(
                false, afr[0], false, bfr, (short)0, acc[0][nt], false, false);
            acc[1][nt] = __builtin_amdgcn_wmma_f32_16x16x32_bf16(
                false, afr[1], false, bfr, (short)0, acc[1][nt], false, false);
        }
    }

    // epilogue: + bias + positional encoding -> sE (f32)
    #pragma unroll
    for (int mt = 0; mt < 2; ++mt)
        #pragma unroll
        for (int nt = 0; nt < 4; ++nt) {
            int N = wv * 64 + nt * 16 + (lane & 15);
            #pragma unroll
            for (int i = 0; i < 8; ++i) {
                int M = 16 * mt + ((lane < 16) ? i : i + 8);
                if (M < TLEN) {
                    sE[M * DMODEL + N] = acc[mt][nt][i] + inconv_b[N]
                                       + pe[(b * TLEN + M) * DHEAD + (N & 15)];
                }
            }
        }
    __syncthreads();

    // ---------- GEMM2: K = E(24x256) @ Wk^T(256x64), wave owns one 16-wide N-tile ----------
    v8f acc2[2];
    { v8f z = {}; acc2[0] = z; acc2[1] = z; }

    #pragma unroll
    for (int kt = 0; kt < 8; ++kt) {
        if (kt < 7)
            __builtin_prefetch(wsBk + (size_t)((kt + 1) * 4 + wv) * 512, 0, 0);

        v16bf afr[2];
        #pragma unroll
        for (int mt = 0; mt < 2; ++mt) {
            int Mrow = (lane & 15) + 16 * mt;
            int hi   = (lane >= 16) ? 16 : 0;
            float f[16];
            if (Mrow < TLEN) {
                const float4* r0 = reinterpret_cast<const float4*>(&sE[Mrow * DMODEL + kt * 32 + hi]);
                const float4* r1 = reinterpret_cast<const float4*>(&sE[Mrow * DMODEL + kt * 32 + 8 + hi]);
                float4 u0 = r0[0], u1 = r0[1], u2 = r1[0], u3 = r1[1];
                f[0]=u0.x; f[1]=u0.y; f[2]=u0.z; f[3]=u0.w;
                f[4]=u1.x; f[5]=u1.y; f[6]=u1.z; f[7]=u1.w;
                f[8]=u2.x; f[9]=u2.y; f[10]=u2.z; f[11]=u2.w;
                f[12]=u3.x; f[13]=u3.y; f[14]=u3.z; f[15]=u3.w;
            } else {
                #pragma unroll
                for (int i = 0; i < 16; ++i) f[i] = 0.0f;
            }
            #pragma unroll
            for (int i = 0; i < 16; ++i) afr[mt][i] = (__bf16)f[i];
        }

        v16bf bfr = *reinterpret_cast<const v16bf*>(wsBk + (size_t)(kt * 4 + wv) * 512 + lane * 16);
        acc2[0] = __builtin_amdgcn_wmma_f32_16x16x32_bf16(
            false, afr[0], false, bfr, (short)0, acc2[0], false, false);
        acc2[1] = __builtin_amdgcn_wmma_f32_16x16x32_bf16(
            false, afr[1], false, bfr, (short)0, acc2[1], false, false);
    }

    // epilogue: K -> sK (overlays sX; all sX readers synced above)
    #pragma unroll
    for (int mt = 0; mt < 2; ++mt) {
        int N = wv * 16 + (lane & 15);
        #pragma unroll
        for (int i = 0; i < 8; ++i) {
            int M = 16 * mt + ((lane < 16) ? i : i + 8);
            if (M < TLEN) sK[M * 64 + N] = acc2[mt][i] + fc1k_b[N];
        }
    }
    __syncthreads();

    // ---------- attention: one learned query per head, softmax over T ----------
    if (tid < NHEAD) {
        const int h = tid;
        const float sc = 0.5f;            // 1/sqrt(D_K)
        float q0 = Qw[h * 4 + 0] * sc, q1 = Qw[h * 4 + 1] * sc;
        float q2 = Qw[h * 4 + 2] * sc, q3 = Qw[h * 4 + 3] * sc;
        float lg[TLEN];
        float mx = -1e30f;
        #pragma unroll
        for (int t = 0; t < TLEN; ++t) {
            const float* kr = &sK[t * 64 + h * 4];
            float v = q0 * kr[0] + q1 * kr[1] + q2 * kr[2] + q3 * kr[3];
            if (pad_mask[b * TLEN + t]) v = -1000.0f;
            lg[t] = v;
            mx = fmaxf(mx, v);
        }
        float sum = 0.f;
        #pragma unroll
        for (int t = 0; t < TLEN; ++t) {
            float e = expf(lg[t] - mx);
            sAttn[t * NHEAD + h] = e;
            sum += e;
        }
        float inv = 1.0f / sum;
        #pragma unroll
        for (int t = 0; t < TLEN; ++t) sAttn[t * NHEAD + h] *= inv;
    }
    __syncthreads();

    // ---------- heads: weighted sum of V (= E channel-split) ----------
    for (int o = tid; o < DMODEL; o += 128) {
        int h = o >> 4;
        float s = 0.f;
        #pragma unroll
        for (int t = 0; t < TLEN; ++t)
            s += sAttn[t * NHEAD + h] * sE[t * DMODEL + o];
        sHeads[o] = s;
    }
    __syncthreads();

    // ---------- proj: (128x256) matvec, coalesced via transposed weights ----------
    {
        int m = tid;                      // 0..127
        float a = proj_b[m];
        #pragma unroll 8
        for (int d = 0; d < DMODEL; ++d)
            a += projT[d * MLPOUT + m] * sHeads[d];
        out[(size_t)p * MLPOUT + m] = a;
    }
}

// =====================================================================
extern "C" void kernel_launch(void* const* d_in, const int* in_sizes, int n_in,
                              void* d_out, int out_size, void* d_ws, size_t ws_size,
                              hipStream_t stream) {
    (void)in_sizes; (void)n_in; (void)out_size; (void)ws_size;

    const float*         x        = (const float*)d_in[0];
    const int*           pos      = (const int*)d_in[1];
    const unsigned char* pad_mask = (const unsigned char*)d_in[2];
    const float*         ln_w     = (const float*)d_in[3];
    const float*         ln_b     = (const float*)d_in[4];
    const float*         inconv_w = (const float*)d_in[5];
    const float*         inconv_b = (const float*)d_in[6];
    const float*         Qw       = (const float*)d_in[7];
    const float*         fc1k_w   = (const float*)d_in[8];
    const float*         fc1k_b   = (const float*)d_in[9];
    const float*         proj_w   = (const float*)d_in[10];
    const float*         proj_b   = (const float*)d_in[11];

    char* ws = (char*)d_ws;
    __bf16* wsBin = (__bf16*)(ws + WS_BIN_OFF);
    __bf16* wsBk  = (__bf16*)(ws + WS_BK_OFF);
    float*  projT = (float*)(ws + WS_PROJT_OFF);
    float*  peTab = (float*)(ws + WS_PE_OFF);

    const int prepWork = 32768 + 16384 + 32768 + SZB * TLEN * DHEAD;
    ltae_prep<<<(prepWork + 255) / 256, 256, 0, stream>>>(
        inconv_w, fc1k_w, proj_w, pos, wsBin, wsBk, projT, peTab);

    ltae_fused<<<SZB * LPIX, 128, 0, stream>>>(
        x, pad_mask, ln_w, ln_b, inconv_b, Qw, fc1k_b, proj_b,
        wsBin, wsBk, projT, peTab, (float*)d_out);
}
